// MultiHeadAttention_81037442941649
// MI455X (gfx1250) — matrix operand order repaired and checked by reading, not compiled, not run on hardware
//
#include <hip/hip_runtime.h>
#include <hip/hip_bf16.h>
#include <stdint.h>

// Multi-head attention, B=4, S=2048, D=PROJ=256, H=8, dh=32.
// All GEMMs via v_wmma_f32_16x16x32_bf16 (bf16 in, f32 accumulate).

#define BATCH   4
#define SEQ     2048
#define DMODEL  256
#define NHEADS  8
#define DHEAD   32
#define PROJ    256
#define SEQW    (SEQ / 32)        // mask words per row

typedef __attribute__((ext_vector_type(16))) __bf16 bf16x16;
typedef __attribute__((ext_vector_type(8)))  float  f32x8;

union AFrag { bf16x16 v; uint4 q[2]; unsigned short u[16]; };
union Acc   { f32x8 v; float f[8]; };

// hardware bf16 convert (lowered to v_cvt_pk_bf16_f32 where pairable; RNE)
__device__ __forceinline__ unsigned short f2bf(float x) {
  __bf16 h = (__bf16)x;
  return __builtin_bit_cast(unsigned short, h);
}

__device__ __forceinline__ f32x8 zero8() {
  f32x8 z = {0.f, 0.f, 0.f, 0.f, 0.f, 0.f, 0.f, 0.f};
  return z;
}

__device__ __forceinline__ f32x8 wmma_bf16(bf16x16 a, bf16x16 b, f32x8 c) {
  return __builtin_amdgcn_wmma_f32_16x16x32_bf16(
      /*neg_a=*/false, a, /*neg_b=*/false, b,
      /*c_mod=*/(short)0, c, /*reuse_a=*/false, /*reuse_b=*/false);
}

// A operand (16x32, M x K), bf16 row-major source.
// Lane r(=lane&15) holds row r; K chunks {hi*8..hi*8+7} and {16+hi*8..+7}.
__device__ __forceinline__ bf16x16 load_a_frag(const unsigned short* base, int stride,
                                               int row, int k0, int lane) {
  const int r = lane & 15, hi = lane >> 4;
  const unsigned short* p = base + (size_t)(row + r) * stride + k0 + hi * 8;
  AFrag f;
  f.q[0] = *(const uint4*)p;
  f.q[1] = *(const uint4*)(p + 16);
  return f.v;
}

// A operand built from fp32 source, converted to bf16 on the fly.
__device__ __forceinline__ bf16x16 load_a_frag_f32(const float* base, int stride,
                                                   int row, int k0, int lane) {
  const int r = lane & 15, hi = lane >> 4;
  const float* p = base + (size_t)(row + r) * stride + k0 + hi * 8;
  float4 a0 = *(const float4*)(p);
  float4 a1 = *(const float4*)(p + 4);
  float4 b0 = *(const float4*)(p + 16);
  float4 b1 = *(const float4*)(p + 20);
  AFrag f;
  f.u[0]  = f2bf(a0.x); f.u[1]  = f2bf(a0.y); f.u[2]  = f2bf(a0.z); f.u[3]  = f2bf(a0.w);
  f.u[4]  = f2bf(a1.x); f.u[5]  = f2bf(a1.y); f.u[6]  = f2bf(a1.z); f.u[7]  = f2bf(a1.w);
  f.u[8]  = f2bf(b0.x); f.u[9]  = f2bf(b0.y); f.u[10] = f2bf(b0.z); f.u[11] = f2bf(b0.w);
  f.u[12] = f2bf(b1.x); f.u[13] = f2bf(b1.y); f.u[14] = f2bf(b1.z); f.u[15] = f2bf(b1.w);
  return f.v;
}

// B operand (32x16, K x N). Source is "B-transposed" row-major: row n holds B[:,n].
// Lane n(=lane&15) holds column n; K chunks {hi*8..hi*8+7} and {16+hi*8..+7}.
__device__ __forceinline__ bf16x16 load_b_frag(const unsigned short* base, int stride,
                                               int col, int k0, int lane) {
  const int n = lane & 15, hi = lane >> 4;
  const unsigned short* p = base + (size_t)(col + n) * stride + k0 + hi * 8;
  AFrag f;
  f.q[0] = *(const uint4*)p;
  f.q[1] = *(const uint4*)(p + 16);
  return f.v;
}

// ---------------------------------------------------------------------------
// Kernel 1: convert the four weight matrices to bf16, transposed.
// ---------------------------------------------------------------------------
__global__ __launch_bounds__(256) void wconv_kernel(const float* __restrict__ Wq,
                                                    const float* __restrict__ Wk,
                                                    const float* __restrict__ Wv,
                                                    const float* __restrict__ Wo,
                                                    unsigned short* __restrict__ ws) {
  int idx = blockIdx.x * blockDim.x + threadIdx.x;   // 4 * 65536 threads
  int w = idx >> 16;
  int n = (idx >> 8) & 255;
  int k = idx & 255;
  const float* src = (w == 0) ? Wq : (w == 1) ? Wk : (w == 2) ? Wv : Wo;
  ws[(size_t)w * 65536 + n * 256 + k] = f2bf(src[k * 256 + n]);
}

// ---------------------------------------------------------------------------
// Kernel 1b: pack the bool mask (bytes are 0/1) into bitmask words.
// bits[b][q][kw] bit j = mask[b][q][kw*32+j]
// ---------------------------------------------------------------------------
__global__ __launch_bounds__(256) void mpack_kernel(const unsigned char* __restrict__ mask,
                                                    unsigned int* __restrict__ bits) {
  int idx = blockIdx.x * blockDim.x + threadIdx.x;   // B*S*S/32 = 524288
  const unsigned int* p = (const unsigned int*)(mask + (size_t)idx * 32);
  unsigned int out = 0;
#pragma unroll
  for (int j = 0; j < 8; j++) {
    unsigned int u = p[j];
    unsigned int nib = (u & 1u) | ((u >> 7) & 2u) | ((u >> 14) & 4u) | ((u >> 21) & 8u);
    out |= nib << (4 * j);
  }
  bits[idx] = out;
}

// ---------------------------------------------------------------------------
// Kernel 2: QKV projections. blockIdx.y selects which of {Q,K,V}.
// Each wave computes a 16x64 output tile (4 WMMA accumulators share one A frag).
// Q,K stored head-split [B,H,S,32] bf16; V stored transposed [B,H,32,S] bf16.
// ---------------------------------------------------------------------------
__global__ __launch_bounds__(256) void proj_qkv_kernel(
    const float* __restrict__ Xq, const float* __restrict__ Xk, const float* __restrict__ Xv,
    const unsigned short* __restrict__ WtQ, const unsigned short* __restrict__ WtK,
    const unsigned short* __restrict__ WtV,
    const float* __restrict__ bq, const float* __restrict__ bk, const float* __restrict__ bv,
    unsigned short* __restrict__ Qb, unsigned short* __restrict__ Kb,
    unsigned short* __restrict__ Vt) {
  const int lane = threadIdx.x & 31, wave = threadIdx.x >> 5;
  const int which = blockIdx.y;                 // 0=Q 1=K 2=V
  const int idx = blockIdx.x * 8 + wave;        // 0..2047 wave-tiles
  const int ng = idx & 3, mt = idx >> 2;        // 4 N-tiles per wave, mt 0..511

  const float* X = (which == 0) ? Xq : (which == 1) ? Xk : Xv;
  const unsigned short* Wt = (which == 0) ? WtQ : (which == 1) ? WtK : WtV;
  const float* bias = (which == 0) ? bq : (which == 1) ? bk : bv;

  Acc acc[4];
#pragma unroll
  for (int j = 0; j < 4; j++) acc[j].v = zero8();

#pragma unroll
  for (int k0 = 0; k0 < DMODEL; k0 += 32) {
    bf16x16 a = load_a_frag_f32(X, DMODEL, mt * 16, k0, lane);
#pragma unroll
    for (int j = 0; j < 4; j++) {
      bf16x16 bf = load_b_frag(Wt, DMODEL, (ng * 4 + j) * 16, k0, lane);
      acc[j].v = wmma_bf16(a, bf, acc[j].v);
    }
  }

  const int hi = lane >> 4;
#pragma unroll
  for (int j = 0; j < 4; j++) {
    const int n = (ng * 4 + j) * 16 + (lane & 15);  // global output column
    const int h = n >> 5, d = n & 31;
    const float bb = bias[n];
#pragma unroll
    for (int e = 0; e < 8; e++) {
      int row = mt * 16 + e + hi * 8;           // global row in [0, B*S)
      int b = row >> 11, s = row & (SEQ - 1);
      unsigned short val = f2bf(acc[j].f[e] + bb);
      if (which == 2) {
        Vt[(((size_t)(b * NHEADS + h)) * DHEAD + d) * SEQ + s] = val;
      } else {
        unsigned short* dst = (which == 0) ? Qb : Kb;
        dst[(((size_t)(b * NHEADS + h)) * SEQ + s) * DHEAD + d] = val;
      }
    }
  }
}

// ---------------------------------------------------------------------------
// Kernel 3: flash attention, block-cooperative.
// 8 waves/block share one (b,h); K/V chunks staged in LDS (double-buffered).
// Per 32-key chunk: 2 score WMMAs + 2 PV WMMAs per wave.
// ---------------------------------------------------------------------------
__global__ __launch_bounds__(256) void attn_kernel(
    const unsigned short* __restrict__ Qb, const unsigned short* __restrict__ Kb,
    const unsigned short* __restrict__ Vt, const unsigned int* __restrict__ Mbits,
    unsigned short* __restrict__ Ob) {
  __shared__ unsigned short Ks[2][32 * 32];     // K chunk [32 keys][32 dh]
  __shared__ unsigned short Vs[2][32 * 32];     // V chunk [32 dh][32 keys]
  __shared__ unsigned short Pl[8][16 * 32];     // per-wave P staging (bf16)

  const int t = threadIdx.x;
  const int lane = t & 31, wave = t >> 5;
  const int bh = blockIdx.x >> 4;               // 0..31  (b*8+h)
  const int qg = blockIdx.x & 15;               // q-tile group
  const int qt = qg * 8 + wave;                 // this wave's q-tile (16 rows)
  const int b = bh >> 3;

  const unsigned short* Qh = Qb + (size_t)bh * SEQ * DHEAD;
  const unsigned short* Kh = Kb + (size_t)bh * SEQ * DHEAD;
  const unsigned short* Vh = Vt + (size_t)bh * DHEAD * SEQ;
  const unsigned int*   Mq = Mbits + ((size_t)b * SEQ + qt * 16) * SEQW;
  unsigned short* Pw = &Pl[wave][0];

  const int r = lane & 15, hi = lane >> 4;
  const float scale = 0.0625f;                  // 1/sqrt(PROJ) = 1/16

  bf16x16 qa = load_a_frag(Qh, DHEAD, qt * 16, 0, lane);

  float mrow[8], lrow[8];
  Acc o0, o1;
#pragma unroll
  for (int e = 0; e < 8; e++) { mrow[e] = -1e30f; lrow[e] = 0.f; o0.f[e] = 0.f; o1.f[e] = 0.f; }

  // cooperative stage of chunk at kc into buffer buf (t 0..127: K, 128..255: V)
  auto stage = [&](int kc, int buf) {
    if (t < 128) {
      int row = t >> 2, part = t & 3;
      *(uint4*)&Ks[buf][row * 32 + part * 8] =
          *(const uint4*)(Kh + (size_t)(kc + row) * DHEAD + part * 8);
    } else {
      int t2 = t - 128;
      int d = t2 >> 2, part = t2 & 3;
      *(uint4*)&Vs[buf][d * 32 + part * 8] =
          *(const uint4*)(Vh + (size_t)d * SEQ + kc + part * 8);
    }
  };

  stage(0, 0);
  __syncthreads();

  for (int i = 0; i < SEQ / 32; i++) {
    const int buf = i & 1;

    if (i + 1 < SEQ / 32) stage((i + 1) * 32, buf ^ 1);
    if (i + 2 < SEQ / 32) {                     // prefetch 2 chunks ahead into caches
      int kc2 = (i + 2) * 32;
      if (t < 16)
        __builtin_prefetch((const void*)(Kh + (size_t)kc2 * DHEAD + t * 64), 0, 0);
      else if (t >= 64 && t < 96)
        __builtin_prefetch((const void*)(Vh + (size_t)(t - 64) * SEQ + kc2), 0, 0);
    }

    // scores: S[16q, 32k] = Q @ K^T ; B operand cols = rows of K (from LDS)
    bf16x16 kb0 = load_b_frag(&Ks[buf][0], 32, 0,  0, lane);
    bf16x16 kb1 = load_b_frag(&Ks[buf][0], 32, 16, 0, lane);
    Acc c0, c1;
    c0.v = wmma_bf16(qa, kb0, zero8());
    c1.v = wmma_bf16(qa, kb1, zero8());

#pragma unroll
    for (int e = 0; e < 8; e++) {
      // C layout: local row = e + hi*8, col = r (tile0) / 16+r (tile1)
      unsigned int mw = Mq[(size_t)(e + hi * 8) * SEQW + i];
      float mk0 = (float)((mw >> r) & 1u);
      float mk1 = (float)((mw >> (16 + r)) & 1u);
      float s0 = (mk0 != 0.f) ? c0.f[e] * scale : -1e9f;
      float s1 = (mk1 != 0.f) ? c1.f[e] * scale : -1e9f;

      float cm = fmaxf(s0, s1);                 // row max over 16 lanes
#pragma unroll
      for (int off = 1; off < 16; off <<= 1) cm = fmaxf(cm, __shfl_xor(cm, off, 32));

      float mn = fmaxf(mrow[e], cm);
      float alpha = __expf(mrow[e] - mn);
      float p0 = mk0 * __expf(s0 - mn);         // exact zero where masked
      float p1 = mk1 * __expf(s1 - mn);
      float rs = p0 + p1;
#pragma unroll
      for (int off = 1; off < 16; off <<= 1) rs += __shfl_xor(rs, off, 32);

      lrow[e] = lrow[e] * alpha + rs;
      mrow[e] = mn;
      o0.f[e] *= alpha;
      o1.f[e] *= alpha;

      Pw[(e + hi * 8) * 32 + r]      = f2bf(p0);
      Pw[(e + hi * 8) * 32 + 16 + r] = f2bf(p1);
    }
    asm volatile("" ::: "memory");              // keep per-wave ds stores before ds loads

    bf16x16 pa;                                 // re-read P as A fragment (16x32)
    {
      AFrag f;
      const unsigned short* p = Pw + r * 32 + hi * 8;
      f.q[0] = *(const uint4*)p;
      f.q[1] = *(const uint4*)(p + 16);
      pa = f.v;
    }

    // PV: B operand cols = rows of Vs (dh-major, from LDS)
    bf16x16 vb0 = load_b_frag(&Vs[buf][0], 32, 0,  0, lane);
    bf16x16 vb1 = load_b_frag(&Vs[buf][0], 32, 16, 0, lane);
    o0.v = wmma_bf16(pa, vb0, o0.v);
    o1.v = wmma_bf16(pa, vb1, o1.v);

    __syncthreads();                            // staging(i+1) done; compute(i) done
  }

  // epilogue: normalize and store merged [B,S,PROJ] bf16
  const int h = bh & 7;
#pragma unroll
  for (int e = 0; e < 8; e++) {
    float inv = 1.0f / fmaxf(lrow[e], 1e-37f);
    int srow = qt * 16 + e + hi * 8;
    size_t rowbase = ((size_t)b * SEQ + srow) * PROJ + h * DHEAD;
    Ob[rowbase + r]      = f2bf(o0.f[e] * inv);
    Ob[rowbase + 16 + r] = f2bf(o1.f[e] * inv);
  }
}

// ---------------------------------------------------------------------------
// Kernel 4: output projection O[8192,256] @ Wo + bo -> fp32 out
// Each wave computes a 16x64 output tile (4 accumulators share one A frag).
// ---------------------------------------------------------------------------
__global__ __launch_bounds__(256) void oproj_kernel(
    const unsigned short* __restrict__ Ob, const unsigned short* __restrict__ WtO,
    const float* __restrict__ bo, float* __restrict__ out) {
  const int lane = threadIdx.x & 31, wave = threadIdx.x >> 5;
  const int idx = blockIdx.x * 8 + wave;        // 0..2047
  const int ng = idx & 3, mt = idx >> 2;

  Acc acc[4];
#pragma unroll
  for (int j = 0; j < 4; j++) acc[j].v = zero8();

#pragma unroll
  for (int k0 = 0; k0 < PROJ; k0 += 32) {
    bf16x16 a = load_a_frag(Ob, PROJ, mt * 16, k0, lane);
#pragma unroll
    for (int j = 0; j < 4; j++) {
      bf16x16 bf = load_b_frag(WtO, PROJ, (ng * 4 + j) * 16, k0, lane);
      acc[j].v = wmma_bf16(a, bf, acc[j].v);
    }
  }

  const int hi = lane >> 4;
#pragma unroll
  for (int j = 0; j < 4; j++) {
    const int n = (ng * 4 + j) * 16 + (lane & 15);
    const float bb = bo[n];
#pragma unroll
    for (int e = 0; e < 8; e++) {
      int row = mt * 16 + e + hi * 8;
      out[(size_t)row * DMODEL + n] = acc[j].f[e] + bb;
    }
  }
}

// ---------------------------------------------------------------------------
extern "C" void kernel_launch(void* const* d_in, const int* in_sizes, int n_in,
                              void* d_out, int out_size, void* d_ws, size_t ws_size,
                              hipStream_t stream) {
  const float* Xq = (const float*)d_in[0];
  const float* Xk = (const float*)d_in[1];
  const float* Xv = (const float*)d_in[2];
  const unsigned char* mask = (const unsigned char*)d_in[3];   // jnp bool = 1 byte
  const float* Wq = (const float*)d_in[4];
  const float* bq = (const float*)d_in[5];
  const float* Wk = (const float*)d_in[6];
  const float* bk = (const float*)d_in[7];
  const float* Wv = (const float*)d_in[8];
  const float* bv = (const float*)d_in[9];
  const float* Wo = (const float*)d_in[10];
  const float* bo = (const float*)d_in[11];

  // workspace (bf16 elems): 4 transposed weights + Q + K + Vt + O, then mask bits
  unsigned short* ws  = (unsigned short*)d_ws;
  unsigned short* WtQ = ws;
  unsigned short* WtK = WtQ + 65536;
  unsigned short* WtV = WtK + 65536;
  unsigned short* WtO = WtV + 65536;
  unsigned short* Qb  = WtO + 65536;
  const size_t NBS = (size_t)BATCH * SEQ * PROJ;   // 2,097,152
  unsigned short* Kb  = Qb + NBS;
  unsigned short* Vt  = Kb + NBS;
  unsigned short* Ob  = Vt + NBS;
  unsigned int*   Mb  = (unsigned int*)(Ob + NBS); // B*S*S/32 words = 2 MB
  // total ws use ~= 18.6 MB

  wconv_kernel<<<dim3(1024), dim3(256), 0, stream>>>(Wq, Wk, Wv, Wo, ws);
  mpack_kernel<<<dim3((BATCH * SEQ * SEQ / 32) / 256), dim3(256), 0, stream>>>(mask, Mb);
  proj_qkv_kernel<<<dim3(256, 3), dim3(256), 0, stream>>>(
      Xq, Xk, Xv, WtQ, WtK, WtV, bq, bk, bv, Qb, Kb, Vt);
  attn_kernel<<<dim3(512), dim3(256), 0, stream>>>(Qb, Kb, Vt, Mb, Ob);
  oproj_kernel<<<dim3(256), dim3(256), 0, stream>>>(Ob, WtO, bo, (float*)d_out);
}
